// LSTMCell_53970559041914
// MI455X (gfx1250) — matrix operand order repaired and checked by reading, not compiled, not run on hardware
//
#include <hip/hip_runtime.h>
#include <stdint.h>

// ---- problem constants (from the reference) ----
#define B_SZ 8192
#define D_SZ 1024
#define U_SZ 1024
#define K_SZ 2048   // D + U
#define G_SZ 4096   // 4 * U

// ---- tile config: 256 threads = 8 waves (4x2); block = 128 rows x 64 units x 4 gates
#define BM 128
#define BN 64
#define BK 32
#define NKB (K_SZ / BK)

typedef __attribute__((ext_vector_type(16))) __bf16         v16bf;
typedef __attribute__((ext_vector_type(8)))  float          v8f;
typedef __attribute__((ext_vector_type(4)))  float          f32x4;
typedef __attribute__((ext_vector_type(4)))  unsigned int   u32x4;
typedef __attribute__((ext_vector_type(4)))  unsigned short u16x4;

__device__ __forceinline__ unsigned short f32_to_bf16_rne(float f) {
    unsigned u = __builtin_bit_cast(unsigned, f);
    u += 0x7FFFu + ((u >> 16) & 1u);
    return (unsigned short)(u >> 16);
}
__device__ __forceinline__ float bf16_as_f32(unsigned short h) {
    unsigned u = ((unsigned)h) << 16;
    return __builtin_bit_cast(float, u);
}
__device__ __forceinline__ void split_bf16(float f, unsigned short& hi, unsigned short& lo) {
    hi = f32_to_bf16_rne(f);
    lo = f32_to_bf16_rne(f - bf16_as_f32(hi));
}

struct U16x16 { u32x4 a, b; };
__device__ __forceinline__ v16bf ld_frag(const unsigned short* p0, const unsigned short* p1) {
    U16x16 t;
    t.a = *(const u32x4*)p0;
    t.b = *(const u32x4*)p1;
    return __builtin_bit_cast(v16bf, t);
}

__device__ __forceinline__ float sigmoidf_fast(float x) {
    return 1.0f / (1.0f + __expf(-x));
}

__global__ __launch_bounds__(256)
void lstm_cell_wmma(const float* __restrict__ x,
                    const float* __restrict__ hprev,
                    const float* __restrict__ cprev,
                    const float* __restrict__ W,
                    const float* __restrict__ bias,
                    float* __restrict__ out_h,
                    float* __restrict__ out_c)
{
    // Padded LDS: A rows stride 40 shorts (80B, mult. of 16), B rows stride 72 shorts (144B).
    __shared__ __align__(16) unsigned short As_hi[BM][40];
    __shared__ __align__(16) unsigned short As_lo[BM][40];
    __shared__ __align__(16) unsigned short Bs_hi[4][BK][72];
    __shared__ __align__(16) unsigned short Bs_lo[4][BK][72];

    const int tid  = threadIdx.x;
    const int lane = tid & 31;
    const int wave = tid >> 5;     // 0..7
    const int wr   = wave >> 1;    // wave row (0..3) -> 32 rows each
    const int wc   = wave & 1;     // wave col (0..1) -> 32 units each
    const int sel  = lane >> 4;    // half-wave select
    const int nl   = lane & 15;

    const int unitBase = blockIdx.x * BN;   // 0..1023
    const int rowBase  = blockIdx.y * BM;   // 0..8191

    v8f acc[4][2][2];
    #pragma unroll
    for (int g = 0; g < 4; ++g)
        #pragma unroll
        for (int mt = 0; mt < 2; ++mt)
            #pragma unroll
            for (int nt = 0; nt < 2; ++nt)
                acc[g][mt][nt] = (v8f){0.f,0.f,0.f,0.f,0.f,0.f,0.f,0.f};

    // Per-thread staging decomposition (256 threads):
    //   A tile: 128x32 fp32 = 1024 float4 -> 4/thread
    //   B tile: 32 x (4 gates x 64) fp32 = 2048 float4 -> 8/thread
    int aRow[4], aCol[4];
    #pragma unroll
    for (int j = 0; j < 4; ++j) {
        int idx = tid + j * 256;
        aRow[j] = idx >> 3;
        aCol[j] = (idx & 7) << 2;
    }
    int bRow[8], bGate[8], bCol[8];
    #pragma unroll
    for (int j = 0; j < 8; ++j) {
        int idx = tid + j * 256;
        bRow[j]  = idx >> 6;
        bGate[j] = (idx & 63) >> 4;
        bCol[j]  = (idx & 15) << 2;
    }

    f32x4 aReg[4], bReg[8];

    // ---- prologue: prefetch k-tile 0 into registers ----
    {
        const float* asrc = x;  // K0 = 0 < D_SZ
        #pragma unroll
        for (int j = 0; j < 4; ++j)
            aReg[j] = *(const f32x4*)(asrc + (size_t)(rowBase + aRow[j]) * D_SZ + aCol[j]);
        const float* wsrc = W + unitBase;
        #pragma unroll
        for (int j = 0; j < 8; ++j)
            bReg[j] = *(const f32x4*)(wsrc + (size_t)bRow[j] * G_SZ + bGate[j] * U_SZ + bCol[j]);
    }

    for (int kb = 0; kb < NKB; ++kb) {
        // ---- stage current tile (registers -> split bf16 hi/lo -> LDS) ----
        #pragma unroll
        for (int j = 0; j < 4; ++j) {
            f32x4 v = aReg[j];
            unsigned short h0,h1,h2,h3,l0,l1,l2,l3;
            split_bf16(v.x, h0, l0); split_bf16(v.y, h1, l1);
            split_bf16(v.z, h2, l2); split_bf16(v.w, h3, l3);
            *(u16x4*)&As_hi[aRow[j]][aCol[j]] = (u16x4){h0,h1,h2,h3};
            *(u16x4*)&As_lo[aRow[j]][aCol[j]] = (u16x4){l0,l1,l2,l3};
        }
        #pragma unroll
        for (int j = 0; j < 8; ++j) {
            f32x4 v = bReg[j];
            unsigned short h0,h1,h2,h3,l0,l1,l2,l3;
            split_bf16(v.x, h0, l0); split_bf16(v.y, h1, l1);
            split_bf16(v.z, h2, l2); split_bf16(v.w, h3, l3);
            *(u16x4*)&Bs_hi[bGate[j]][bRow[j]][bCol[j]] = (u16x4){h0,h1,h2,h3};
            *(u16x4*)&Bs_lo[bGate[j]][bRow[j]][bCol[j]] = (u16x4){l0,l1,l2,l3};
        }
        __syncthreads();

        // ---- prefetch next k-tile while this one computes ----
        if (kb + 1 < NKB) {
            const int K0n = (kb + 1) * BK;
            const float* asrc = (K0n < D_SZ) ? (x + K0n) : (hprev + (K0n - D_SZ));
            #pragma unroll
            for (int j = 0; j < 4; ++j)
                aReg[j] = *(const f32x4*)(asrc + (size_t)(rowBase + aRow[j]) * D_SZ + aCol[j]);
            const float* wsrc = W + (size_t)K0n * G_SZ + unitBase;
            #pragma unroll
            for (int j = 0; j < 8; ++j)
                bReg[j] = *(const f32x4*)(wsrc + (size_t)bRow[j] * G_SZ + bGate[j] * U_SZ + bCol[j]);
        }

        // ---- A fragments (16-bit A 16x32: lane=M, VGPRs pack K, half-wave K+8) ----
        v16bf ahi[2], alo[2];
        #pragma unroll
        for (int mt = 0; mt < 2; ++mt) {
            const unsigned short* ph = &As_hi[wr * 32 + mt * 16 + nl][sel * 8];
            const unsigned short* pl = &As_lo[wr * 32 + mt * 16 + nl][sel * 8];
            ahi[mt] = ld_frag(ph, ph + 16);   // K = sel*8+0..7 and 16+sel*8+0..7
            alo[mt] = ld_frag(pl, pl + 16);
        }

        // ---- B fragments (lane = N: nl; elements = K sel*8+0..7 and 16+sel*8+0..7) + WMMA ----
        #pragma unroll
        for (int g = 0; g < 4; ++g) {
            #pragma unroll
            for (int nt = 0; nt < 2; ++nt) {
                const int bcol = wc * 32 + nt * 16 + nl;
                v16bf bhi, blo;
                #pragma unroll
                for (int s = 0; s < 16; ++s) {
                    const int kk = sel * 8 + (s & 7) + (s >> 3) * 16;
                    bhi[s] = __builtin_bit_cast(__bf16, Bs_hi[g][kk][bcol]);
                    blo[s] = __builtin_bit_cast(__bf16, Bs_lo[g][kk][bcol]);
                }
                #pragma unroll
                for (int mt = 0; mt < 2; ++mt) {
                    v8f a = acc[g][mt][nt];
                    // split product: hi*hi + hi*lo + lo*hi (lo*lo dropped, ~2^-16 rel)
                    a = __builtin_amdgcn_wmma_f32_16x16x32_bf16(false, ahi[mt], false, bhi, (short)0, a, false, false);
                    a = __builtin_amdgcn_wmma_f32_16x16x32_bf16(false, ahi[mt], false, blo, (short)0, a, false, false);
                    a = __builtin_amdgcn_wmma_f32_16x16x32_bf16(false, alo[mt], false, bhi, (short)0, a, false, false);
                    acc[g][mt][nt] = a;
                }
            }
        }
        __syncthreads();
    }

    // ---- fused LSTM epilogue (C/D layout: lane=N, VGPR v = M row, half-wave +8) ----
    #pragma unroll
    for (int nt = 0; nt < 2; ++nt) {
        const int u = unitBase + wc * 32 + nt * 16 + nl;
        const float bi = bias[u];
        const float bf = bias[U_SZ + u];
        const float bo = bias[2 * U_SZ + u];
        const float bg = bias[3 * U_SZ + u];
        #pragma unroll
        for (int mt = 0; mt < 2; ++mt) {
            #pragma unroll
            for (int v = 0; v < 8; ++v) {
                const int r = rowBase + wr * 32 + mt * 16 + sel * 8 + v;
                const size_t off = (size_t)r * U_SZ + u;
                const float gi = acc[0][mt][nt][v] + bi;   // i
                const float gf = acc[1][mt][nt][v] + bf;   // f
                const float go = acc[2][mt][nt][v] + bo;   // o
                const float gg = acc[3][mt][nt][v] + bg;   // g
                const float cc = cprev[off] * sigmoidf_fast(gf) + tanhf(gg) * sigmoidf_fast(gi);
                const float hh = sigmoidf_fast(go) * tanhf(cc);
                out_h[off] = hh;
                out_c[off] = cc;
            }
        }
    }
}

extern "C" void kernel_launch(void* const* d_in, const int* in_sizes, int n_in,
                              void* d_out, int out_size, void* d_ws, size_t ws_size,
                              hipStream_t stream) {
    (void)in_sizes; (void)n_in; (void)out_size; (void)d_ws; (void)ws_size;
    const float* x  = (const float*)d_in[0];
    const float* hp = (const float*)d_in[1];
    const float* cp = (const float*)d_in[2];
    const float* W  = (const float*)d_in[3];
    const float* b  = (const float*)d_in[4];
    float* out_h = (float*)d_out;                      // h: [8192,1024]
    float* out_c = out_h + (size_t)B_SZ * U_SZ;        // c: [8192,1024]

    dim3 grid(U_SZ / BN, B_SZ / BM);                   // (16, 64)
    dim3 block(256);
    hipLaunchKernelGGL(lstm_cell_wmma, grid, block, 0, stream,
                       x, hp, cp, W, b, out_h, out_c);
}
